// AttentionRoPE_5746666242642
// MI455X (gfx1250) — compile-verified, matrix operand order checked by reading, not run on hardware
//
#include <hip/hip_runtime.h>

typedef unsigned short u16;
typedef __attribute__((ext_vector_type(16))) __bf16 v16bf;
typedef __attribute__((ext_vector_type(8)))  float  v8f;
typedef __attribute__((ext_vector_type(4)))  unsigned v4u;
typedef __attribute__((ext_vector_type(8)))  unsigned v8u;

struct B32x8 { uint4 lo, hi; };

static __device__ __forceinline__ v16bf frag_cast(uint4 lo, uint4 hi) {
    B32x8 t; t.lo = lo; t.hi = hi;
    return __builtin_bit_cast(v16bf, t);
}

// float -> bf16 bits, round-to-nearest-even
static __device__ __forceinline__ u16 f2bf(float f) {
    unsigned u = __builtin_bit_cast(unsigned, f);
    u += 0x7FFFu + ((u >> 16) & 1u);
    return (u16)(u >> 16);
}

// ---------------- Tensor Data Mover: DMA a 2-D bf16 tile (rows x cols, row stride in elems)
// from global memory into LDS at byte offset ldsOff. Issue once per workgroup (gate on one wave);
// completion tracked with TENSORcnt. D# layout per CDNA5 ISA ch.8 (group0 128b + group1 256b).
static __device__ __forceinline__ void tdm_load_tile(const u16* gptr, unsigned ldsOff,
                                                     unsigned rows, unsigned cols,
                                                     unsigned strideElems) {
    unsigned long long ga = (unsigned long long)gptr;
    unsigned galo = __builtin_amdgcn_readfirstlane((unsigned)ga);
    unsigned gahi = __builtin_amdgcn_readfirstlane((unsigned)(ga >> 32));
    unsigned lofs = __builtin_amdgcn_readfirstlane(ldsOff);

    v4u g0;
    g0[0] = 1u;                                   // count=1 (valid user descriptor)
    g0[1] = lofs;                                 // lds_addr (bytes)
    g0[2] = galo;                                 // global_addr[31:0]
    g0[3] = (gahi & 0x01FFFFFFu) | (2u << 30);    // global_addr[56:32] | type=2 (image)

    v8u g1;
    g1[0] = 1u << 16;                             // data_size=1 -> 2 bytes/elem
    g1[1] = (strideElems & 0xFFFFu) << 16;        // tensor_dim0[15:0]  (>= cols, no OOB)
    g1[2] = (strideElems >> 16) | (0xFFFFu << 16);// tensor_dim0[31:16] | tensor_dim1[15:0]=65535
    g1[3] = (cols << 16);                         // tensor_dim1[31:16]=0 | tile_dim0=cols
    g1[4] = rows;                                 // tile_dim1=rows, tile_dim2=0
    g1[5] = strideElems;                          // tensor_dim0_stride[31:0] (elems)
    g1[6] = 0u;                                   // stride hi | tensor_dim1_stride lo
    g1[7] = 0u;

    asm volatile("tensor_load_to_lds %0, %1" : : "s"(g0), "s"(g1) : "memory");
}

// ---------------- conversion kernels ----------------

__global__ void cvt_bf16(const float* __restrict__ s, u16* __restrict__ d, int n) {
    int i = (blockIdx.x * blockDim.x + threadIdx.x) * 4;
    if (i + 3 < n) {
        float4 v = *(const float4*)(s + i);
        d[i + 0] = f2bf(v.x); d[i + 1] = f2bf(v.y);
        d[i + 2] = f2bf(v.z); d[i + 3] = f2bf(v.w);
    }
}

// src fp32 [R][C] row-major -> dst bf16 [C][R]  (pre-transpose weights for B-frags)
__global__ void cvt_t_bf16(const float* __restrict__ s, u16* __restrict__ d, int R, int C) {
    int i = blockIdx.x * blockDim.x + threadIdx.x;
    if (i < R * C) {
        int r = i / C, c = i % C;
        d[(size_t)c * R + r] = f2bf(s[i]);
    }
}

// ---------------- bf16 WMMA GEMM with TDM double-buffered pipeline ----------------
// C[M][N] = A[M][K] * Bt[N][K]^T + bias.  Block = 256 thr (8 waves), tile 128x128, K-stage 64.

__global__ __launch_bounds__(256) void gemm_bf16_wmma(
        const u16* __restrict__ A, const u16* __restrict__ Bt,
        const float* __restrict__ bias, float* __restrict__ C,
        int M, int N, int K) {
    __shared__ __align__(128) u16 as[2][128 * 64];   // 32 KB
    __shared__ __align__(128) u16 bs[2][128 * 64];   // 32 KB

    const int t = threadIdx.x;
    const int wid = t >> 5, lane = t & 31;
    const int wm = wid >> 2, wn = wid & 3;            // 2 x 4 waves -> 64x32 per wave
    const int m0 = blockIdx.y * 128, n0 = blockIdx.x * 128;
    const int mlane = lane & 15, hlf = lane >> 4;

    v8f acc[4][2];
    const v8f vzero = {0.f, 0.f, 0.f, 0.f, 0.f, 0.f, 0.f, 0.f};
#pragma unroll
    for (int mt = 0; mt < 4; ++mt)
#pragma unroll
        for (int nt = 0; nt < 2; ++nt) acc[mt][nt] = vzero;

    const int nstage = K >> 6;
    const unsigned asOff0 = (unsigned)(unsigned long long)&as[0][0];
    const unsigned asOff1 = (unsigned)(unsigned long long)&as[1][0];
    const unsigned bsOff0 = (unsigned)(unsigned long long)&bs[0][0];
    const unsigned bsOff1 = (unsigned)(unsigned long long)&bs[1][0];

    if (wid == 0) {   // prologue: DMA stage 0 into buffer 0
        tdm_load_tile(&A[(size_t)m0 * K + 0], asOff0, 128, 64, (unsigned)K);
        tdm_load_tile(&Bt[(size_t)n0 * K + 0], bsOff0, 128, 64, (unsigned)K);
    }

    for (int s = 0; s < nstage; ++s) {
        const int buf = s & 1;
        if (wid == 0) {
            if (s + 1 < nstage) {   // prefetch next stage into the other buffer
                const int k1 = (s + 1) << 6;
                tdm_load_tile(&A[(size_t)m0 * K + k1], buf ? asOff0 : asOff1, 128, 64, (unsigned)K);
                tdm_load_tile(&Bt[(size_t)n0 * K + k1], buf ? bsOff0 : bsOff1, 128, 64, (unsigned)K);
                __builtin_amdgcn_s_wait_tensorcnt((short)2);  // current stage done; prefetch in flight
            } else {
                __builtin_amdgcn_s_wait_tensorcnt((short)0);
            }
        }
        __syncthreads();   // TDM data visible to all waves

        const u16* ab = &as[buf][0];
        const u16* bb = &bs[buf][0];
#pragma unroll
        for (int k2 = 0; k2 < 2; ++k2) {            // two 32-deep WMMA steps per stage
            v16bf af[4], bfm[2];
#pragma unroll
            for (int mt = 0; mt < 4; ++mt) {
                const u16* p = &ab[(wm * 64 + mt * 16 + mlane) * 64 + k2 * 32];
                af[mt] = frag_cast(*(const uint4*)&p[hlf * 8],
                                   *(const uint4*)&p[16 + hlf * 8]);
            }
#pragma unroll
            for (int nt = 0; nt < 2; ++nt) {
                const u16* p = &bb[(wn * 32 + nt * 16 + mlane) * 64 + k2 * 32];
                bfm[nt] = frag_cast(*(const uint4*)&p[hlf * 8],
                                    *(const uint4*)&p[16 + hlf * 8]);
            }
#pragma unroll
            for (int mt = 0; mt < 4; ++mt)
#pragma unroll
                for (int nt = 0; nt < 2; ++nt)
                    acc[mt][nt] = __builtin_amdgcn_wmma_f32_16x16x32_bf16(
                        false, af[mt], false, bfm[nt], (short)0, acc[mt][nt], false, false);
        }
        __syncthreads();   // all waves done with buf before it is DMA-overwritten
    }

#pragma unroll
    for (int mt = 0; mt < 4; ++mt)
#pragma unroll
        for (int nt = 0; nt < 2; ++nt)
#pragma unroll
            for (int r = 0; r < 8; ++r) {
                int row = m0 + wm * 64 + mt * 16 + hlf * 8 + r;
                int col = n0 + wn * 32 + nt * 16 + mlane;
                C[(size_t)row * N + col] = acc[mt][nt][r] + bias[col];
            }
}

// ---------------- RoPE + pack:  qkv fp32 [B*N][3*1024] -> Qh,Kh [B,H,N,64] bf16 ; Vt [B,H,64,N] bf16 ----------------

__global__ void rope_pack(const float* __restrict__ qkv,
                          u16* __restrict__ Qh, u16* __restrict__ Kh, u16* __restrict__ Vt) {
    int idx = blockIdx.x * blockDim.x + threadIdx.x;   // 2*2048*16*32 threads
    int i = idx & 31;
    int h = (idx >> 5) & 15;
    int n = (idx >> 9) & 2047;
    int b = idx >> 20;

    size_t base = (size_t)(b * 2048 + n) * 3072;
    float q0 = qkv[base + h * 64 + 2 * i];
    float q1 = qkv[base + h * 64 + 2 * i + 1];
    float k0 = qkv[base + 1024 + h * 64 + 2 * i];
    float k1 = qkv[base + 1024 + h * 64 + 2 * i + 1];
    float v0 = qkv[base + 2048 + h * 64 + 2 * i];
    float v1 = qkv[base + 2048 + h * 64 + 2 * i + 1];

    // theta^(-2i/64) = exp(-i * ln(10000)/32)
    float freq = __expf(-(float)i * (9.2103403720f / 32.0f));
    float ang = (float)n * freq;
    float s, c;
    __sincosf(ang, &s, &c);
    const float sc = 0.125f;                            // fold in HEAD_DIM^-0.5
    float qr = (q0 * c - q1 * s) * sc, qi = (q0 * s + q1 * c) * sc;
    float kr = k0 * c - k1 * s,        ki = k0 * s + k1 * c;

    size_t qb = ((size_t)(b * 16 + h) * 2048 + n) * 64 + 2 * i;
    Qh[qb] = f2bf(qr); Qh[qb + 1] = f2bf(qi);
    Kh[qb] = f2bf(kr); Kh[qb + 1] = f2bf(ki);
    size_t vb = ((size_t)(b * 16 + h) * 64 + 2 * i) * 2048 + n;
    Vt[vb] = f2bf(v0); Vt[vb + 2048] = f2bf(v1);
}

// ---------------- flash attention: per (b,h), 64-query tile, TDM-staged K/V, WMMA everywhere ----------------

__global__ __launch_bounds__(128) void flash_attn(
        const u16* __restrict__ Qh, const u16* __restrict__ Kh,
        const u16* __restrict__ Vt, u16* __restrict__ outh) {
    constexpr int Nk = 2048, HD = 64;
    __shared__ __align__(128) u16 ks[64 * 64];
    __shared__ __align__(128) u16 vs[64 * 64];
    __shared__ __align__(16)  u16 ps[4 * 16 * 64];       // per-wave P scratch

    const int t = threadIdx.x, wid = t >> 5, lane = t & 31;
    const int mlane = lane & 15, hlf = lane >> 4;
    const int qt = blockIdx.x, bh = blockIdx.y;

    const u16* Qb = Qh + (size_t)bh * Nk * HD;
    const u16* Kb = Kh + (size_t)bh * Nk * HD;
    const u16* Vb = Vt + (size_t)bh * HD * Nk;
    const unsigned ksOff = (unsigned)(unsigned long long)&ks[0];
    const unsigned vsOff = (unsigned)(unsigned long long)&vs[0];

    // Q fragments (16 rows per wave, K-dim = head_dim = 64 -> 2 frags)
    v16bf aq[2];
    {
        const u16* qp = Qb + (size_t)(qt * 64 + wid * 16 + mlane) * HD;
#pragma unroll
        for (int k2 = 0; k2 < 2; ++k2)
            aq[k2] = frag_cast(*(const uint4*)&qp[k2 * 32 + hlf * 8],
                               *(const uint4*)&qp[k2 * 32 + 16 + hlf * 8]);
    }

    const v8f vzero = {0.f, 0.f, 0.f, 0.f, 0.f, 0.f, 0.f, 0.f};
    v8f oacc[4];
#pragma unroll
    for (int nt = 0; nt < 4; ++nt) oacc[nt] = vzero;
    float mrun[8], lrun[8];
#pragma unroll
    for (int r = 0; r < 8; ++r) { mrun[r] = -3.0e38f; lrun[r] = 0.f; }

    for (int kt = 0; kt < Nk / 64; ++kt) {
        if (wid == 0) {   // DMA K tile (contiguous) and V^T tile (strided rows) into LDS
            tdm_load_tile(Kb + (size_t)(kt * 64) * HD, ksOff, 64, 64, HD);
            tdm_load_tile(Vb + kt * 64,               vsOff, 64, 64, Nk);
            __builtin_amdgcn_s_wait_tensorcnt((short)0);
        }
        __syncthreads();

        // S = Q * K^T  (16 q-rows x 64 keys per wave)
        v8f sacc[4];
#pragma unroll
        for (int nt = 0; nt < 4; ++nt) sacc[nt] = vzero;
#pragma unroll
        for (int nt = 0; nt < 4; ++nt)
#pragma unroll
            for (int k2 = 0; k2 < 2; ++k2) {
                const u16* p = &ks[(nt * 16 + mlane) * 64 + k2 * 32];
                v16bf bk = frag_cast(*(const uint4*)&p[hlf * 8],
                                     *(const uint4*)&p[16 + hlf * 8]);
                sacc[nt] = __builtin_amdgcn_wmma_f32_16x16x32_bf16(
                    false, aq[k2], false, bk, (short)0, sacc[nt], false, false);
            }

        // online softmax (rows live per-VGPR, cols per-lane within 16-lane group)
        float mnew[8], alpha[8], rs[8];
#pragma unroll
        for (int r = 0; r < 8; ++r) {
            float v = fmaxf(fmaxf(sacc[0][r], sacc[1][r]), fmaxf(sacc[2][r], sacc[3][r]));
            v = fmaxf(v, __shfl_xor(v, 1, 32));
            v = fmaxf(v, __shfl_xor(v, 2, 32));
            v = fmaxf(v, __shfl_xor(v, 4, 32));
            v = fmaxf(v, __shfl_xor(v, 8, 32));
            mnew[r] = fmaxf(mrun[r], v);
            alpha[r] = __expf(mrun[r] - mnew[r]);
            mrun[r] = mnew[r];
            rs[r] = 0.f;
        }
#pragma unroll
        for (int nt = 0; nt < 4; ++nt)
#pragma unroll
            for (int r = 0; r < 8; ++r) {
                float p = __expf(sacc[nt][r] - mnew[r]);
                rs[r] += p;
                ps[wid * 1024 + (hlf * 8 + r) * 64 + nt * 16 + mlane] = f2bf(p);
            }
#pragma unroll
        for (int r = 0; r < 8; ++r) {
            float v = rs[r];
            v += __shfl_xor(v, 1, 32);
            v += __shfl_xor(v, 2, 32);
            v += __shfl_xor(v, 4, 32);
            v += __shfl_xor(v, 8, 32);
            lrun[r] = lrun[r] * alpha[r] + v;
        }
#pragma unroll
        for (int nt = 0; nt < 4; ++nt)
#pragma unroll
            for (int r = 0; r < 8; ++r) oacc[nt][r] *= alpha[r];

        asm volatile("s_wait_dscnt 0" ::: "memory");    // P stores visible before re-read (same wave)

        // O += P * V
#pragma unroll
        for (int k2 = 0; k2 < 2; ++k2) {
            const u16* pp = &ps[wid * 1024 + mlane * 64 + k2 * 32];
            v16bf ap = frag_cast(*(const uint4*)&pp[hlf * 8],
                                 *(const uint4*)&pp[16 + hlf * 8]);
#pragma unroll
            for (int nt = 0; nt < 4; ++nt) {
                const u16* vp = &vs[(nt * 16 + mlane) * 64 + k2 * 32];
                v16bf bv = frag_cast(*(const uint4*)&vp[hlf * 8],
                                     *(const uint4*)&vp[16 + hlf * 8]);
                oacc[nt] = __builtin_amdgcn_wmma_f32_16x16x32_bf16(
                    false, ap, false, bv, (short)0, oacc[nt], false, false);
            }
        }
        __syncthreads();
    }

    // write attention output as bf16 in [B*N][H*64] layout for the proj GEMM
    const int b = bh >> 4, h = bh & 15;
#pragma unroll
    for (int r = 0; r < 8; ++r) {
        float inv = 1.0f / lrun[r];
#pragma unroll
        for (int nt = 0; nt < 4; ++nt) {
            int row = b * Nk + qt * 64 + wid * 16 + hlf * 8 + r;
            int col = h * 64 + nt * 16 + mlane;
            outh[(size_t)row * 1024 + col] = f2bf(oacc[nt][r] * inv);
        }
    }
}

// ---------------- host launch ----------------

extern "C" void kernel_launch(void* const* d_in, const int* in_sizes, int n_in,
                              void* d_out, int out_size, void* d_ws, size_t ws_size,
                              hipStream_t stream) {
    const float* x     = (const float*)d_in[0];   // [2,2048,1024]
    const float* Wqkv  = (const float*)d_in[1];   // [1024,3072]
    const float* bqkv  = (const float*)d_in[2];   // [3072]
    const float* Wproj = (const float*)d_in[3];   // [1024,1024]
    const float* bproj = (const float*)d_in[4];   // [1024]
    (void)in_sizes; (void)n_in; (void)out_size; (void)ws_size;

    char* ws = (char*)d_ws;
    constexpr size_t MB = 1u << 20;
    u16*   xh     = (u16*)(ws + 0);               //  8 MB  x bf16 [4096][1024]
    u16*   wqkvt  = (u16*)(ws + 8  * MB);         //  6 MB  Wqkv^T bf16 [3072][1024]
    u16*   wprojt = (u16*)(ws + 14 * MB);         //  2 MB  Wproj^T bf16 [1024][1024]
    float* qkvf   = (float*)(ws + 16 * MB);       // 48 MB  qkv fp32 [4096][3072]
    u16*   Qhp    = (u16*)(ws + 64 * MB);         //  8 MB  [B,H,N,64]
    u16*   Khp    = (u16*)(ws + 72 * MB);         //  8 MB  [B,H,N,64]
    u16*   Vtp    = (u16*)(ws + 80 * MB);         //  8 MB  [B,H,64,N]
    u16*   outh   = (u16*)(ws + 88 * MB);         //  8 MB  attn out bf16 [4096][1024]

    cvt_bf16<<<4096, 256, 0, stream>>>(x, xh, 4194304);
    cvt_t_bf16<<<12288, 256, 0, stream>>>(Wqkv, wqkvt, 1024, 3072);
    cvt_t_bf16<<<4096, 256, 0, stream>>>(Wproj, wprojt, 1024, 1024);

    gemm_bf16_wmma<<<dim3(24, 32), 256, 0, stream>>>(xh, wqkvt, bqkv, qkvf, 4096, 3072, 1024);

    rope_pack<<<8192, 256, 0, stream>>>(qkvf, Qhp, Khp, Vtp);

    flash_attn<<<dim3(32, 32), 128, 0, stream>>>(Qhp, Khp, Vtp, outh);

    gemm_bf16_wmma<<<dim3(8, 32), 256, 0, stream>>>(outh, wprojt, bproj, (float*)d_out,
                                                    4096, 1024, 1024);
}
